// LocalFeatureLossLayer_86938728005658
// MI455X (gfx1250) — matrix operand order repaired and checked by reading, not compiled, not run on hardware
//
#include <hip/hip_runtime.h>
#include <hip/hip_fp16.h>

typedef __attribute__((ext_vector_type(16))) _Float16 v16h;
typedef __attribute__((ext_vector_type(8)))  _Float16 v8h;
typedef __attribute__((ext_vector_type(8)))  float    v8f;

#define WAVES 8
#define BLOCK 256

// Force global (addrspace(1)) vector loads so f16 operand fetches lower to
// global_load_b128 (LOADcnt only) instead of flat_load_b128 (LOADcnt+DScnt,
// which false-couples WMMA waits to the LDS counter and burns the LDS path).
typedef const __attribute__((address_space(1))) v8h* gv8p;
__device__ __forceinline__ v8h gload8(const _Float16* p) {
  return *(gv8p)p;
}

__device__ __forceinline__ v8f zero_v8f() {
  v8f z;
#pragma unroll
  for (int i = 0; i < 8; ++i) z[i] = 0.0f;
  return z;
}

__device__ __forceinline__ v16h cat8(v8h lo, v8h hi) {
  v16h r;
#pragma unroll
  for (int i = 0; i < 8; ++i) { r[i] = lo[i]; r[i + 8] = hi[i]; }
  return r;
}

__device__ __forceinline__ float redmax16(float v) {
#pragma unroll
  for (int m = 1; m < 16; m <<= 1) v = fmaxf(v, __shfl_xor(v, m, 32));
  return v;
}
__device__ __forceinline__ float redsum16(float v) {
#pragma unroll
  for (int m = 1; m < 16; m <<= 1) v += __shfl_xor(v, m, 32);
  return v;
}
__device__ __forceinline__ float redsum32(float v) {
#pragma unroll
  for (int m = 1; m < 32; m <<= 1) v += __shfl_xor(v, m, 32);
  return v;
}

// ---------------------------------------------------------------------------
// Instance-norm statistics: per (batch, channel) mean & rsqrt(var+eps) over N
// spatial positions. One thread per channel -> fully coalesced loads.
// ---------------------------------------------------------------------------
__global__ void inorm_stats_kernel(const float* __restrict__ x,
                                   float* __restrict__ mean,
                                   float* __restrict__ rstd,
                                   int N, int CH, float eps) {
  int ch = blockIdx.x * blockDim.x + threadIdx.x;
  int b  = blockIdx.y;
  if (ch >= CH) return;
  const float* p = x + (size_t)b * N * CH + ch;
  float s = 0.f, s2 = 0.f;
  for (int n = 0; n < N; ++n) {
    float v = p[(size_t)n * CH];
    s += v; s2 += v * v;
  }
  float m   = s / (float)N;
  float var = fmaxf(s2 / (float)N - m * m, 0.f);
  mean[b * CH + ch] = m;
  rstd[b * CH + ch] = rsqrtf(var + eps);
}

// Normalize and convert to f16 (row-major [B][N][CH]) for WMMA Q/K operands.
__global__ void norm_f16_kernel(const float* __restrict__ x,
                                const float* __restrict__ mean,
                                const float* __restrict__ rstd,
                                _Float16* __restrict__ out,
                                int N, int CH, int total) {
  int idx = blockIdx.x * blockDim.x + threadIdx.x;
  if (idx >= total) return;
  int ch = idx % CH;
  int b  = (idx / CH) / N;
  float v = (x[idx] - mean[b * CH + ch]) * rstd[b * CH + ch];
  out[idx] = (_Float16)v;
}

// Build transposed value matrix Vt[B][2C][N] = [v ; v*v] in f16, so WMMA-B
// operands (per-lane contiguous along K = key index) are direct 16B loads.
__global__ void build_vt_kernel(const float* __restrict__ s,
                                _Float16* __restrict__ Vt,
                                int N, int C, int total) {
  int idx = blockIdx.x * blockDim.x + threadIdx.x;
  if (idx >= total) return;
  int ch = idx % C;
  int bn = idx / C;
  int n  = bn % N;
  int b  = bn / N;
  float v = s[idx];
  size_t base = (size_t)b * 2 * C * N;
  Vt[base + (size_t)ch * N + n]       = (_Float16)v;
  Vt[base + (size_t)(C + ch) * N + n] = (_Float16)(v * v);
}

// ---------------------------------------------------------------------------
// Fused flash-attention AAT kernel, templated on TM = C/128 (2 or 4) so the
// P @ [V ; V^2] accumulation is a straight line of 8*TM WMMAs per 128-key
// step. Grid: (N/16, B); block: 256 threads = 8 wave32. Each block owns 16
// query rows; per 128-key step each wave produces one 16x16 score tile
// (v_wmma_f32_16x16x32_f16 over Cc), the block does cross-wave online
// softmax, then all waves consume the shared exp'd P tile against their
// private column chunk of [V ; V^2].
// ---------------------------------------------------------------------------
template <int TM>
__global__ __launch_bounds__(BLOCK)
void aat_flash_kernel(const _Float16* __restrict__ Q16,
                      const _Float16* __restrict__ K16,
                      const _Float16* __restrict__ Vt,
                      const float* __restrict__ c,
                      const float* __restrict__ cs,
                      const float* __restrict__ mean_c,
                      const float* __restrict__ rstd_c,
                      float* __restrict__ partial,
                      int N, int Cc, float inv_total) {
  constexpr int C   = TM * 128;  // content channels
  constexpr int CMW = TM * 16;   // columns per wave per stream
  extern __shared__ char smem_raw[];
  _Float16* Qs = (_Float16*)smem_raw;        // 16 x Cc   (query rows)
  _Float16* Ps = Qs + 16 * Cc;               // 16 x 128  (exp'd probabilities)
  float* wmax  = (float*)(Ps + 16 * 128);    // WAVES x 16 per-wave row maxima
  float* wsum  = wmax + WAVES * 16;          // WAVES x 16 per-wave row sums

  const int tid  = threadIdx.x;
  const int wave = tid >> 5;
  const int lane = tid & 31;
  const int col  = lane & 15;   // column within 16x16 tile
  const int hi   = lane >> 4;   // lane half: rows 0-7 vs 8-15 for C/D; K split
  const int b    = blockIdx.y;
  const int row0 = blockIdx.x * 16;

  // Stage the 16 contiguous Q rows into LDS (16*Cc halfs, 16B chunks).
  {
    const int nchunk = (16 * Cc) >> 3;
    const uint4* src = (const uint4*)(Q16 + ((size_t)b * N + row0) * Cc);
    uint4* dst = (uint4*)Qs;
    for (int i = tid; i < nchunk; i += BLOCK) dst[i] = src[i];
  }
  __syncthreads();

  float rowmax[8], rowsum[8];
#pragma unroll
  for (int i = 0; i < 8; ++i) { rowmax[i] = -3.0e38f; rowsum[i] = 0.f; }

  v8f accM[TM], accS[TM];
#pragma unroll
  for (int t = 0; t < TM; ++t) { accM[t] = zero_v8f(); accS[t] = zero_v8f(); }

  const size_t kbase = (size_t)b * N * Cc;

  // Per-tile V / V^2 column pointers: bumped by 128 halfs per key step so the
  // hot-loop addresses are pointer + constant instruction offset.
  const _Float16* vptrM[TM];
  const _Float16* vptrS[TM];
#pragma unroll
  for (int t = 0; t < TM; ++t) {
    int colM = wave * CMW + t * 16 + col;
    const _Float16* base = Vt + ((size_t)b * 2 * C + colM) * N + hi * 16;
    vptrM[t] = base;
    vptrS[t] = base + (size_t)C * N;
  }

  for (int j0 = 0; j0 < N; j0 += 128) {
    // ---- score tile: rows row0..row0+15 vs keys [j0+wave*16, +16) ----
    v8f sAcc = zero_v8f();
    const int key = j0 + wave * 16 + col;
    const _Float16* kptr = K16 + kbase + (size_t)key * Cc + hi * 16;
    const _Float16* qrow = Qs + col * Cc + hi * 16;
    if (j0 + 128 < N) {  // warm next K rows and V chunk (global_prefetch)
      __builtin_prefetch(K16 + kbase + (size_t)(key + 128) * Cc, 0, 1);
      __builtin_prefetch(vptrM[0] + 128, 0, 1);
      __builtin_prefetch(vptrS[0] + 128, 0, 1);
    }
#pragma unroll 2
    for (int kk = 0; kk < Cc; kk += 32) {
      v16h a  = cat8(*(const v8h*)(qrow + kk), *(const v8h*)(qrow + kk + 8));
      v16h bk = cat8(gload8(kptr + kk), gload8(kptr + kk + 8));
      sAcc = __builtin_amdgcn_wmma_f32_16x16x32_f16(
          false, a, false, bk, (short)0, sAcc, false, false);
    }

    __syncthreads();  // previous step's consumers of Ps/wmax/wsum are done

    // per-row maxima of this wave's tile -> LDS for cross-wave combine
#pragma unroll
    for (int i = 0; i < 8; ++i) {
      float tmax = redmax16(sAcc[i]);
      if (col == 0) wmax[wave * 16 + hi * 8 + i] = tmax;
    }
    __syncthreads();

    float nmax[8];
#pragma unroll
    for (int i = 0; i < 8; ++i) {
      float nm = rowmax[i];
      int r = hi * 8 + i;
#pragma unroll
      for (int w = 0; w < WAVES; ++w) nm = fmaxf(nm, wmax[w * 16 + r]);
      nmax[i] = nm;
      float alpha = __expf(rowmax[i] - nm);
      rowmax[i] = nm;
      rowsum[i] *= alpha;
#pragma unroll
      for (int t = 0; t < TM; ++t) { accM[t][i] *= alpha; accS[t][i] *= alpha; }
    }

    // P = exp(s - rowmax) -> f16 LDS tile; per-wave row sums -> LDS
#pragma unroll
    for (int i = 0; i < 8; ++i) {
      float p = __expf(sAcc[i] - nmax[i]);
      Ps[(hi * 8 + i) * 128 + wave * 16 + col] = (_Float16)p;
      float ts = redsum16(p);
      if (col == 0) wsum[wave * 16 + hi * 8 + i] = ts;
    }
    __syncthreads();

#pragma unroll
    for (int i = 0; i < 8; ++i) {
      int r = hi * 8 + i;
      float ssum = 0.f;
#pragma unroll
      for (int w = 0; w < WAVES; ++w) ssum += wsum[w * 16 + r];
      rowsum[i] += ssum;
    }

    // ---- accumulate P @ [V ; V^2]: straight-line 8*TM WMMAs ----
    const _Float16* prow = Ps + col * 128 + hi * 16;
#pragma unroll
    for (int kk = 0; kk < 128; kk += 32) {
      v16h aP = cat8(*(const v8h*)(prow + kk), *(const v8h*)(prow + kk + 8));
#pragma unroll
      for (int t = 0; t < TM; ++t) {
        v16h bm  = cat8(gload8(vptrM[t] + kk), gload8(vptrM[t] + kk + 8));
        v16h bs2 = cat8(gload8(vptrS[t] + kk), gload8(vptrS[t] + kk + 8));
        accM[t] = __builtin_amdgcn_wmma_f32_16x16x32_f16(
            false, aP, false, bm, (short)0, accM[t], false, false);
        accS[t] = __builtin_amdgcn_wmma_f32_16x16x32_f16(
            false, aP, false, bs2, (short)0, accS[t], false, false);
      }
    }
#pragma unroll
    for (int t = 0; t < TM; ++t) { vptrM[t] += 128; vptrS[t] += 128; }
  }

  // ---- epilogue: aat = S * norm(c) + M ; accumulate squared error ----
  float lsum = 0.f;
#pragma unroll
  for (int t = 0; t < TM; ++t) {
    int colM = wave * CMW + t * 16 + col;
    float mn = mean_c[b * C + colM];
    float rs = rstd_c[b * C + colM];
#pragma unroll
    for (int i = 0; i < 8; ++i) {
      int row = row0 + hi * 8 + i;
      float inv_l = 1.f / rowsum[i];
      float m  = accM[t][i] * inv_l;
      float e2 = accS[t][i] * inv_l;
      float s2 = e2 - m * m;
      float sd = sqrtf(fmaxf(fmaxf(s2, 0.f), 1e-9f));
      size_t idx = ((size_t)b * N + row) * C + colM;
      float nc = (c[idx] - mn) * rs;
      float d  = cs[idx] - (sd * nc + m);
      lsum += d * d;
    }
  }
  lsum = redsum32(lsum);
  __syncthreads();
  if (lane == 0) wmax[wave] = lsum;
  __syncthreads();
  if (tid == 0) {
    float tot = 0.f;
#pragma unroll
    for (int w = 0; w < WAVES; ++w) tot += wmax[w];
    partial[blockIdx.y * gridDim.x + blockIdx.x] = tot * inv_total;
  }
}

// Deterministic final sum of the pre-scaled per-block partials.
__global__ void reduce_partials_kernel(const float* __restrict__ partial, int n,
                                       float* __restrict__ out) {
  __shared__ float sh[BLOCK];
  float s = 0.f;
  for (int i = threadIdx.x; i < n; i += BLOCK) s += partial[i];
  sh[threadIdx.x] = s;
  __syncthreads();
  for (int st = BLOCK / 2; st > 0; st >>= 1) {
    if (threadIdx.x < st) sh[threadIdx.x] += sh[threadIdx.x + st];
    __syncthreads();
  }
  if (threadIdx.x == 0) out[0] = sh[0];
}

// ---------------------------------------------------------------------------
// Host side
// ---------------------------------------------------------------------------
static void run_scale(const float* cs, const float* c, const float* s,
                      const float* cc, const float* sc,
                      int B, int N, int C, int Cc,
                      char* region, float* partial, hipStream_t stream) {
  size_t off = 0;
  auto carve = [&](size_t bytes) -> char* {
    char* p = region + off;
    off += (bytes + 255) & ~(size_t)255;
    return p;
  };
  _Float16* Q16 = (_Float16*)carve((size_t)B * N * Cc * 2);
  _Float16* K16 = (_Float16*)carve((size_t)B * N * Cc * 2);
  _Float16* Vt  = (_Float16*)carve((size_t)B * 2 * C * N * 2);
  float* mean_c = (float*)carve((size_t)B * C * 4);
  float* rstd_c = (float*)carve((size_t)B * C * 4);
  float* mean_q = (float*)carve((size_t)B * Cc * 4);
  float* rstd_q = (float*)carve((size_t)B * Cc * 4);
  float* mean_k = (float*)carve((size_t)B * Cc * 4);
  float* rstd_k = (float*)carve((size_t)B * Cc * 4);

  dim3 blk(BLOCK);
  inorm_stats_kernel<<<dim3((C + BLOCK - 1) / BLOCK, B), blk, 0, stream>>>(
      c, mean_c, rstd_c, N, C, 1e-5f);
  inorm_stats_kernel<<<dim3((Cc + BLOCK - 1) / BLOCK, B), blk, 0, stream>>>(
      cc, mean_q, rstd_q, N, Cc, 1e-5f);
  inorm_stats_kernel<<<dim3((Cc + BLOCK - 1) / BLOCK, B), blk, 0, stream>>>(
      sc, mean_k, rstd_k, N, Cc, 1e-5f);

  int totq = B * N * Cc;
  norm_f16_kernel<<<(totq + BLOCK - 1) / BLOCK, blk, 0, stream>>>(
      cc, mean_q, rstd_q, Q16, N, Cc, totq);
  norm_f16_kernel<<<(totq + BLOCK - 1) / BLOCK, blk, 0, stream>>>(
      sc, mean_k, rstd_k, K16, N, Cc, totq);

  int totv = B * N * C;
  build_vt_kernel<<<(totv + BLOCK - 1) / BLOCK, blk, 0, stream>>>(
      s, Vt, N, C, totv);

  size_t lds = (size_t)16 * Cc * 2    // Q rows
             + (size_t)16 * 128 * 2   // P tile
             + (size_t)2 * WAVES * 16 * 4;  // wmax + wsum
  float inv_total = 1.0f / ((float)B * (float)N * (float)C);
  if (C == 256) {
    aat_flash_kernel<2><<<dim3(N / 16, B), blk, lds, stream>>>(
        Q16, K16, Vt, c, cs, mean_c, rstd_c, partial, N, Cc, inv_total);
  } else {
    aat_flash_kernel<4><<<dim3(N / 16, B), blk, lds, stream>>>(
        Q16, K16, Vt, c, cs, mean_c, rstd_c, partial, N, Cc, inv_total);
  }
}

extern "C" void kernel_launch(void* const* d_in, const int* in_sizes, int n_in,
                              void* d_out, int out_size, void* d_ws, size_t ws_size,
                              hipStream_t stream) {
  (void)in_sizes; (void)n_in; (void)out_size; (void)ws_size;
  const float* cs2 = (const float*)d_in[0];
  const float* c2  = (const float*)d_in[1];
  const float* s2  = (const float*)d_in[2];
  const float* cc2 = (const float*)d_in[3];
  const float* sc2 = (const float*)d_in[4];
  const float* cs3 = (const float*)d_in[5];
  const float* c3  = (const float*)d_in[6];
  const float* s3  = (const float*)d_in[7];
  const float* cc3 = (const float*)d_in[8];
  const float* sc3 = (const float*)d_in[9];
  const float* cs4 = (const float*)d_in[10];
  const float* c4  = (const float*)d_in[11];
  const float* s4  = (const float*)d_in[12];
  const float* cc4 = (const float*)d_in[13];
  const float* sc4 = (const float*)d_in[14];

  // Workspace: [ partials (1344 f32) | reusable per-scale region ]
  const int NPART = 1024 + 256 + 64;  // flash blocks: scale2 + scale3 + scale4
  float* partial = (float*)d_ws;
  char* region = (char*)d_ws + (((size_t)NPART * 4 + 255) & ~(size_t)255);

  // Scales processed sequentially on one stream -> region reuse is safe.
  run_scale(cs2, c2, s2, cc2, sc2, 4, 4096, 256, 448,  region, partial,        stream);
  run_scale(cs3, c3, s3, cc3, sc3, 4, 1024, 512, 960,  region, partial + 1024, stream);
  run_scale(cs4, c4, s4, cc4, sc4, 4,  256, 512, 1472, region, partial + 1280, stream);

  reduce_partials_kernel<<<1, BLOCK, 0, stream>>>(partial, NPART, (float*)d_out);
}
